// GraphLabelLayer_71837622993134
// MI455X (gfx1250) — matrix operand order repaired and checked by reading, not compiled, not run on hardware
//
#include <hip/hip_runtime.h>

// ---------------------------------------------------------------------------
// Problem constants
// ---------------------------------------------------------------------------
#define NCLS   151              // object classes incl. background
#define CM1    150
#define NPAIR  512              // pairs per image
#define NIMG   128
#define MROWS  (NIMG * NPAIR)   // 65536 output rows
#define KDIM   600              // 4*(NCLS-1)
#define SEGP   160              // each 150-wide segment padded to 160
#define KPAD   (4 * SEGP)       // 640 = 20 chunks of 32
#define NKCH   (KPAD / 32)      // 20
#define NDIM   2048             // PATH_DIM
#define MT     64               // M tile per block
#define NT     128              // N tile per block (8 wmma tiles / wave)

typedef __attribute__((ext_vector_type(16))) _Float16 v16h;
typedef __attribute__((ext_vector_type(8)))  _Float16 v8h;
typedef __attribute__((ext_vector_type(8)))  float    v8f;
typedef __attribute__((ext_vector_type(4)))  int      v4i;

// ---------------------------------------------------------------------------
// gfx1250 async global->LDS copy (16B per lane), with safe fallback.
// Probe-confirmed signature: (int4 AS1* src, int4 AS3* dst, imm, imm)
// ---------------------------------------------------------------------------
#if defined(__AMDGCN__) && __has_builtin(__builtin_amdgcn_global_load_async_to_lds_b128)
#define HAVE_ASYNC_LDS 1
typedef __attribute__((address_space(1))) v4i* gv4i_p;
typedef __attribute__((address_space(3))) v4i* lv4i_p;
#else
#define HAVE_ASYNC_LDS 0
#endif

__device__ __forceinline__ void copy16B_to_lds(const _Float16* __restrict__ gsrc,
                                               _Float16* ldst) {
#if HAVE_ASYNC_LDS
    __builtin_amdgcn_global_load_async_to_lds_b128(
        (gv4i_p)(uintptr_t)gsrc,
        (lv4i_p)(unsigned)(uintptr_t)ldst,
        0, 0);
#else
    *(v8h*)ldst = *(const v8h*)gsrc;
#endif
}

__device__ __forceinline__ void wait_async_copies() {
#if HAVE_ASYNC_LDS
#if __has_builtin(__builtin_amdgcn_s_wait_asynccnt)
    __builtin_amdgcn_s_wait_asynccnt(0);
#else
    asm volatile("s_wait_asynccnt 0x0" ::: "memory");
#endif
#endif
}

// ---------------------------------------------------------------------------
// Prep 1: sym = (score + score^T) * 0.5   [151 x 151] f32
// ---------------------------------------------------------------------------
__global__ void sym_kernel(const float* __restrict__ score, float* __restrict__ sym) {
    int idx = blockIdx.x * blockDim.x + threadIdx.x;
    if (idx >= NCLS * NCLS) return;
    int i = idx / NCLS, j = idx % NCLS;
    sym[idx] = 0.5f * (score[i * NCLS + j] + score[j * NCLS + i]);
}

// ---------------------------------------------------------------------------
// Prep 2: per-image presence mask [128 x 151] f32 (one block per image)
// ---------------------------------------------------------------------------
__global__ void mask_kernel(const int* __restrict__ pairs, float* __restrict__ mask) {
    int b = blockIdx.x, tid = threadIdx.x;
    float* mrow = mask + b * NCLS;
    for (int i = tid; i < NCLS; i += blockDim.x) mrow[i] = 0.0f;
    __syncthreads();
    const int* pp = pairs + b * (NPAIR * 2);
    for (int i = tid; i < NPAIR * 2; i += blockDim.x) {
        int c = pp[i];
        if (c >= 0 && c < NCLS) mrow[c] = 1.0f;   // benign races: all store 1.0
    }
}

// ---------------------------------------------------------------------------
// Prep 3: erow[b][c][j] = sym[c][j+1]*mask[b][j+1], f16, j padded to 160
// ---------------------------------------------------------------------------
__global__ void erow_kernel(const float* __restrict__ sym, const float* __restrict__ mask,
                            _Float16* __restrict__ erow) {
    int idx = blockIdx.x * blockDim.x + threadIdx.x;
    if (idx >= NIMG * NCLS * SEGP) return;
    int j   = idx % SEGP;
    int bc  = idx / SEGP;
    int c   = bc % NCLS;
    int b   = bc / NCLS;
    float v = 0.0f;
    if (j < CM1) v = sym[c * NCLS + j + 1] * mask[b * NCLS + j + 1];
    erow[idx] = (_Float16)v;
}

// ---------------------------------------------------------------------------
// Prep 4: one-hot table oh[c][j] = (j == c-1), f16, padded to 160
// ---------------------------------------------------------------------------
__global__ void oh_kernel(_Float16* __restrict__ oh) {
    int idx = blockIdx.x * blockDim.x + threadIdx.x;
    if (idx >= NCLS * SEGP) return;
    int j = idx % SEGP, c = idx / SEGP;
    oh[idx] = (_Float16)((j < CM1 && j == c - 1) ? 1.0f : 0.0f);
}

// ---------------------------------------------------------------------------
// Prep 5: weights -> f16, n-major [2048][640] with 150->160 segment padding
// ---------------------------------------------------------------------------
__global__ void wcvt_kernel(const float* __restrict__ lin_w, _Float16* __restrict__ wcol) {
    int idx = blockIdx.x * blockDim.x + threadIdx.x;   // idx = n*KPAD + k
    if (idx >= NDIM * KPAD) return;
    int k = idx % KPAD, n = idx / KPAD;
    int s = k / SEGP, j = k - s * SEGP;
    wcol[idx] = (j < CM1) ? (_Float16)lin_w[n * KDIM + s * CM1 + j] : (_Float16)0.0f;
}

// ---------------------------------------------------------------------------
// Main GEMM: out[65536 x 2048] = context @ W^T + b  (WMMA f16 -> f32)
// 128 threads (4 wave32), tile 64(M) x 128(N), 20 K-chunks of 32,
// double-buffered LDS with async global->LDS staging.
// ---------------------------------------------------------------------------
__launch_bounds__(128)
__global__ void gemm_kernel(const int*      __restrict__ pairs,
                            const _Float16* __restrict__ erow,
                            const _Float16* __restrict__ oh,
                            const _Float16* __restrict__ wcol,
                            const float*    __restrict__ lin_b,
                            float*          __restrict__ out) {
    // 80B row stride keeps 16B alignment for ds_load_b128 and avoids conflicts
    __shared__ __align__(16) _Float16 sA [2][MT][40];   // A tile  [row][k]
    __shared__ __align__(16) _Float16 sBT[2][NT][40];   // B tile  [n][k] (col-major)
    __shared__ int sP0[MT], sP1[MT];

    const int tid  = threadIdx.x;
    const int g0   = blockIdx.x * MT;     // first pair-row of tile
    const int n0   = blockIdx.y * NT;     // first output column of tile
    const int bimg = g0 >> 9;             // 64 | 512 -> tile is within one image

    if (tid < MT) {
        int g = g0 + tid;
        sP0[tid] = pairs[g * 2 + 0];
        sP1[tid] = pairs[g * 2 + 1];
    }
    __syncthreads();

    const int w    = tid >> 5;            // wave 0..3 -> M strip 16*w
    const int ln   = tid & 31;
    const int half = ln >> 4;
    const int lc   = ln & 15;

    // ---- staging coordinates (branch-free sources) ----
    const int ar  = tid >> 1;             // A: row 0..63, two threads per row
    const int aks = (tid & 1) * 16;       // A: halfword group 0 / 16
    const int p0  = sP0[ar], p1 = sP1[ar];
    const _Float16* aseg[4];
    aseg[0] = erow + (size_t)(bimg * NCLS + p0) * SEGP;   // e0
    aseg[1] = erow + (size_t)(bimg * NCLS + p1) * SEGP;   // e1
    aseg[2] = oh   + (size_t)p0 * SEGP;                   // t0
    aseg[3] = oh   + (size_t)p1 * SEGP;                   // t1
    const _Float16* bsrc = wcol + (size_t)(n0 + tid) * KPAD;  // B: col = tid

    v8f acc[8];
#pragma unroll
    for (int t = 0; t < 8; ++t) acc[t] = (v8f)0.0f;

    // ---- chunked staging: chunk ch covers K' [ch*32, ch*32+32) in segment ch/5 ----
    auto stage = [&](int buf, int ch) {
        const int s  = ch / 5;                     // block-uniform segment select
        const int ko = (ch - s * 5) * 32;          // offset inside segment
        const _Float16* asrc = aseg[s] + ko + aks;
        copy16B_to_lds(asrc,     &sA[buf][ar][aks]);
        copy16B_to_lds(asrc + 8, &sA[buf][ar][aks + 8]);
        const _Float16* bs = bsrc + ch * 32;
#pragma unroll
        for (int kg = 0; kg < 4; ++kg)
            copy16B_to_lds(bs + kg * 8, &sBT[buf][tid][kg * 8]);
    };

    stage(0, 0);
    wait_async_copies();
    __syncthreads();

    for (int ch = 0; ch < NKCH; ++ch) {
        const int cur = ch & 1;
        if (ch + 1 < NKCH) stage(cur ^ 1, ch + 1);   // prefetch next chunk (async)

        // A fragment: M = lc, K = {half*8+0..7, 16+half*8+0..7}
        const _Float16* ap = &sA[cur][16 * w + lc][half * 8];
        v8h alo = *(const v8h*)ap;
        v8h ahi = *(const v8h*)(ap + 16);
        v16h av = __builtin_shufflevector(alo, ahi, 0, 1, 2, 3, 4, 5, 6, 7,
                                                    8, 9, 10, 11, 12, 13, 14, 15);

        // 8 N-tiles: B fragment N = lc, K = {half*8+0..7, 16+half*8+0..7}
#pragma unroll
        for (int t = 0; t < 8; ++t) {
            const _Float16* bp = &sBT[cur][16 * t + lc][half * 8];
            v8h blo = *(const v8h*)bp;
            v8h bhi = *(const v8h*)(bp + 16);
            v16h bv = __builtin_shufflevector(blo, bhi, 0, 1, 2, 3, 4, 5, 6, 7,
                                                        8, 9, 10, 11, 12, 13, 14, 15);
            acc[t] = __builtin_amdgcn_wmma_f32_16x16x32_f16(
                         false, av, false, bv, (short)0, acc[t], false, false);
        }

        if (ch + 1 < NKCH) wait_async_copies();      // own wave's copies landed
        __syncthreads();                             // publish to all waves
    }

    // ---- epilogue: fused bias, f32 stream-out ----
#pragma unroll
    for (int t = 0; t < 8; ++t) {
        const int col  = n0 + 16 * t + lc;
        const float bb = lin_b[col];
#pragma unroll
        for (int v = 0; v < 8; ++v) {
            const int row = 16 * w + v + 8 * half;   // C/D layout: M = v + 8*(lane/16)
            out[(size_t)(g0 + row) * NDIM + col] = acc[t][v] + bb;
        }
    }
}

// ---------------------------------------------------------------------------
// Launcher
// ---------------------------------------------------------------------------
static inline size_t align256(size_t x) { return (x + 255) & ~(size_t)255; }

extern "C" void kernel_launch(void* const* d_in, const int* in_sizes, int n_in,
                              void* d_out, int out_size, void* d_ws, size_t ws_size,
                              hipStream_t stream) {
    const int*   pairs = (const int*)  d_in[0];   // [128,512,2] int32
    const float* score = (const float*)d_in[1];   // [151,151]   f32
    const float* lin_w = (const float*)d_in[2];   // [2048,600]  f32
    const float* lin_b = (const float*)d_in[3];   // [2048]      f32
    float* out = (float*)d_out;                   // [65536,2048] f32

    char* ws = (char*)d_ws;
    size_t off = 0;
    float*    sym  = (float*)(ws + off);    off = align256(off + (size_t)NCLS * NCLS * 4);
    float*    mask = (float*)(ws + off);    off = align256(off + (size_t)NIMG * NCLS * 4);
    _Float16* oh   = (_Float16*)(ws + off); off = align256(off + (size_t)NCLS * SEGP * 2);
    _Float16* erow = (_Float16*)(ws + off); off = align256(off + (size_t)NIMG * NCLS * SEGP * 2);
    _Float16* wcol = (_Float16*)(ws + off); // [2048][640] f16  (~2.6 MB; total ~9 MB)

    sym_kernel <<<(NCLS * NCLS + 255) / 256,        256, 0, stream>>>(score, sym);
    mask_kernel<<<NIMG,                             256, 0, stream>>>(pairs, mask);
    erow_kernel<<<(NIMG * NCLS * SEGP + 255) / 256, 256, 0, stream>>>(sym, mask, erow);
    oh_kernel  <<<(NCLS * SEGP + 255) / 256,        256, 0, stream>>>(oh);
    wcvt_kernel<<<(NDIM * KPAD + 255) / 256,        256, 0, stream>>>(lin_w, wcol);

    dim3 grid(MROWS / MT, NDIM / NT);   // 1024 x 16
    gemm_kernel<<<grid, 128, 0, stream>>>(pairs, erow, oh, wcol, lin_b, out);
}
